// MoELayer_82566451298774
// MI455X (gfx1250) — matrix operand order repaired and checked by reading, not compiled, not run on hardware
//
#include <hip/hip_runtime.h>

// ---------- types ----------
typedef __attribute__((ext_vector_type(16))) __bf16       v16bf;
typedef __attribute__((ext_vector_type(8)))  float        v8f;
typedef __attribute__((ext_vector_type(4)))  unsigned int u32x4;
typedef __attribute__((ext_vector_type(8)))  int          i32x8;
typedef __attribute__((ext_vector_type(4)))  int          i32x4;

union FragU { u32x4 q[2]; v16bf v; };
union AccU  { v8f v; float f[8]; };

__device__ inline unsigned short f2bf(float f) {
  unsigned u = __float_as_uint(f);
  u += 0x7FFFu + ((u >> 16) & 1u);   // round-to-nearest-even
  return (unsigned short)(u >> 16);
}

// Problem constants
static constexpr int BT = 4096;   // B*T
static constexpr int D  = 1024;
static constexpr int E  = 8;
static constexpr int H  = 4096;

#if defined(__gfx1250__) && __has_builtin(__builtin_amdgcn_tensor_load_to_lds)
#define MOE_USE_TDM 1
#else
#define MOE_USE_TDM 0
#endif

// ---------- gating: logits -> top2 -> softmax -> dense combine weights ----------
__global__ void gate_kernel(const float* __restrict__ x,
                            const float* __restrict__ gw,
                            const float* __restrict__ gb,
                            float* __restrict__ comb) {
  int t = blockIdx.x * blockDim.x + threadIdx.x;   // token id
  if (t >= BT) return;
  float l[E];
#pragma unroll
  for (int e = 0; e < E; ++e) l[e] = gb[e];
  const float* xp = x + (size_t)t * D;
  for (int d = 0; d < D; ++d) {
    float xv = xp[d];
#pragma unroll
    for (int e = 0; e < E; ++e) l[e] += xv * gw[d * E + e];
  }
  int i1 = 0; float v1 = l[0];
#pragma unroll
  for (int e = 1; e < E; ++e) if (l[e] > v1) { v1 = l[e]; i1 = e; }
  int i2 = -1; float v2 = -3.4e38f;
#pragma unroll
  for (int e = 0; e < E; ++e) if (e != i1 && l[e] > v2) { v2 = l[e]; i2 = e; }
  float z  = __expf(v2 - v1);        // softmax over the 2 selected logits
  float p1 = 1.0f / (1.0f + z);
  float p2 = 1.0f - p1;
#pragma unroll
  for (int e = 0; e < E; ++e)
    comb[t * E + e] = (e == i1) ? p1 : ((e == i2) ? p2 : 0.0f);
}

// ---------- x fp32 -> bf16 ----------
__global__ void cvt_bf16_kernel(const float* __restrict__ x,
                                unsigned short* __restrict__ xb, int n) {
  int i = blockIdx.x * blockDim.x + threadIdx.x;
  int stride = gridDim.x * blockDim.x;
  for (; i < n; i += stride) xb[i] = f2bf(x[i]);
}

// ---------- fused WMMA GEMM ----------
// FFN1: Hout[row,col](bf16) = gelu(A@W + bias)      A=[BT,KDIM] bf16, W=[KDIM,NDIM] f32
// FFN2: Out[row,col](f32)  (+)= comb[row,e] * (A@W + bias)
// Block tile 128x128, K-step 32, 8 waves (2x4), each wave 64x32 = 8 WMMA accs.
// A tile staged via TDM (tensor_load_to_lds), B tile converted f32->bf16 in flight.
template<int KDIM, int NDIM, bool FFN1>
__global__ __launch_bounds__(256)
void moe_gemm(const unsigned short* __restrict__ A,
              const float* __restrict__ W,
              const float* __restrict__ bias,
              unsigned short* __restrict__ Hout,
              const float* __restrict__ comb,
              float* __restrict__ Out,
              int expert) {
  __shared__ __align__(16) unsigned short sA[128 * 32];   // row-major [m][k]
  __shared__ __align__(16) unsigned short sB[128 * 32];   // col-major [n][k]

  const int t      = threadIdx.x;
  const int m0     = blockIdx.y * 128;
  const int n0     = blockIdx.x * 128;
  const int lane   = t & 31;
  const int wv     = t >> 5;
  const int lane16 = lane & 15;
  const int hi     = lane >> 4;       // half-wave select
  const int waveM  = wv >> 2;         // 0..1 -> 64-row slab
  const int waveN  = wv & 3;          // 0..3 -> 32-col slab

  const v8f vzero = {0.f, 0.f, 0.f, 0.f, 0.f, 0.f, 0.f, 0.f};
  v8f acc[4][2];
#pragma unroll
  for (int mi = 0; mi < 4; ++mi)
#pragma unroll
    for (int ni = 0; ni < 2; ++ni) acc[mi][ni] = vzero;

  // staging assignments
  const int rowA = t >> 1, halfA = t & 1;   // fallback A copy: 16 bf16 (32B)/thread
  const int nB   = t & 127, kgB = t >> 7;   // B: each thread converts 16 f32 -> 8 packed b32

#if MOE_USE_TDM
  // TDM descriptor invariants (ISA 8.3/8.4): 2D tile 32 x 128 of 2-byte data.
  const unsigned ldsA = (unsigned)(size_t)(&sA[0]);   // low 32 bits of LDS aperture VA
#endif

  for (int k0 = 0; k0 < KDIM; k0 += 32) {
#if MOE_USE_TDM
    // ---- stage A tile via Tensor Data Mover (wave 0 issues one DMA) ----
    if (wv == 0) {
      const unsigned short* gp = A + (size_t)m0 * KDIM + k0;
      unsigned long long ga = (unsigned long long)(size_t)gp;
      u32x4 g0;
      g0.x = 0x1u;                                   // count=1, user descriptor
      g0.y = ldsA;                                   // lds_addr
      g0.z = (unsigned)(ga & 0xFFFFFFFFu);           // global_addr[31:0]
      g0.w = (unsigned)((ga >> 32) & 0x1FFFFFFu)     // global_addr[56:32]
             | (2u << 30);                           // type = 2 ("image")
      i32x8 g1;
      g1[0] = (int)(1u << 16);                       // data_size = 1 -> 2 bytes
      g1[1] = (int)(((unsigned)KDIM & 0xFFFFu) << 16);        // tensor_dim0[15:0]
      g1[2] = (int)((((unsigned)KDIM >> 16) & 0xFFFFu)        // tensor_dim0[31:16]
             | (((unsigned)BT & 0xFFFFu) << 16));             // tensor_dim1[15:0]
      g1[3] = (int)((((unsigned)BT >> 16) & 0xFFFFu)          // tensor_dim1[31:16]
             | (32u << 16));                                  // tile_dim0 = 32 (K)
      g1[4] = 128;                                            // tile_dim1 = 128 rows
      g1[5] = (int)(unsigned)KDIM;                            // tensor_dim0_stride
      g1[6] = 0;
      g1[7] = 0;
      i32x4 gz  = {0, 0, 0, 0};                               // 2D: groups 2/3 unused
      i32x8 gz8 = {0, 0, 0, 0, 0, 0, 0, 0};
      __builtin_amdgcn_tensor_load_to_lds(g0, g1, gz, gz, gz8, 0);
    }
#else
    // ---- stage A tile (bf16 copy, 2 x b128 per thread) ----
    {
      const u32x4* ga = reinterpret_cast<const u32x4*>(
          A + (size_t)(m0 + rowA) * KDIM + k0 + halfA * 16);
      u32x4* da = reinterpret_cast<u32x4*>(&sA[rowA * 32 + halfA * 16]);
      da[0] = ga[0];
      da[1] = ga[1];
    }
#endif

    // ---- stage B tile (f32 -> bf16 pairs, b32 stores, column-major) ----
    if (k0 + 32 < KDIM)  // prefetch next K-step weight rows into cache
      __builtin_prefetch(W + (size_t)(k0 + 32 + kgB) * NDIM + n0 + nB, 0, 1);
#pragma unroll
    for (int i = 0; i < 8; ++i) {
      int k = (kgB * 8 + i) * 2;                     // even k
      float w0 = W[(size_t)(k0 + k) * NDIM + n0 + nB];
      float w1 = W[(size_t)(k0 + k + 1) * NDIM + n0 + nB];
      unsigned pk = (unsigned)f2bf(w0) | ((unsigned)f2bf(w1) << 16);
      *reinterpret_cast<unsigned*>(&sB[nB * 32 + k]) = pk;
    }

#if MOE_USE_TDM
    if (wv == 0) __builtin_amdgcn_s_wait_tensorcnt(0);
#endif
    __syncthreads();

    // ---- load fragments per ISA layouts ----
    v16bf aF[4];
#pragma unroll
    for (int mi = 0; mi < 4; ++mi) {
      int M  = waveM * 64 + mi * 16 + lane16;
      int kb = hi * 8;                 // lanes 0-15: K 0-7/16-23; lanes 16-31: K 8-15/24-31
      FragU u;
      u.q[0] = *reinterpret_cast<const u32x4*>(&sA[M * 32 + kb]);
      u.q[1] = *reinterpret_cast<const u32x4*>(&sA[M * 32 + 16 + kb]);
      aF[mi] = u.v;
    }
    v16bf bF[2];
#pragma unroll
    for (int ni = 0; ni < 2; ++ni) {
      int N   = waveN * 32 + ni * 16 + lane16;
      int klo = hi * 8;                // lanes 0-15: K 0-7/16-23; lanes 16-31: K 8-15/24-31
      FragU u;
      u.q[0] = *reinterpret_cast<const u32x4*>(&sB[N * 32 + klo]);
      u.q[1] = *reinterpret_cast<const u32x4*>(&sB[N * 32 + klo + 16]);
      bF[ni] = u.v;
    }

    // ---- 8 WMMA per wave per K-step ----
#pragma unroll
    for (int mi = 0; mi < 4; ++mi)
#pragma unroll
      for (int ni = 0; ni < 2; ++ni)
        acc[mi][ni] = __builtin_amdgcn_wmma_f32_16x16x32_bf16(
            false, aF[mi], false, bF[ni], (short)0, acc[mi][ni], false, false);

    __syncthreads();
  }

  // ---- epilogue (C/D layout: VGPR g -> M = g + hi*8, N = lane16) ----
#pragma unroll
  for (int mi = 0; mi < 4; ++mi) {
#pragma unroll
    for (int ni = 0; ni < 2; ++ni) {
      AccU a; a.v = acc[mi][ni];
      int col = n0 + waveN * 32 + ni * 16 + lane16;
      float b = bias[col];
#pragma unroll
      for (int g = 0; g < 8; ++g) {
        int row = m0 + waveM * 64 + mi * 16 + hi * 8 + g;
        float v = a.f[g] + b;
        if (FFN1) {
          float gl = 0.5f * v * (1.0f + erff(v * 0.70710678118654752f)); // exact GELU
          Hout[(size_t)row * NDIM + col] = f2bf(gl);
        } else {
          float p   = comb[row * E + expert];
          float val = p * v;
          float* o  = Out + (size_t)row * NDIM + col;
          if (expert == 0) *o = val;     // first expert initializes (d_out is poisoned)
          else             *o = *o + val;
        }
      }
    }
  }
}

// ---------- host launcher ----------
extern "C" void kernel_launch(void* const* d_in, const int* in_sizes, int n_in,
                              void* d_out, int out_size, void* d_ws, size_t ws_size,
                              hipStream_t stream) {
  (void)in_sizes; (void)n_in; (void)out_size; (void)ws_size;
  const float* x  = (const float*)d_in[0];
  const float* gw = (const float*)d_in[1];
  const float* gb = (const float*)d_in[2];
  const float* w1 = (const float*)d_in[3];
  const float* b1 = (const float*)d_in[4];
  const float* w2 = (const float*)d_in[5];
  const float* b2 = (const float*)d_in[6];
  float* out = (float*)d_out;

  // workspace layout (all 16B aligned):
  //   comb : BT*E f32          = 128 KB
  //   xb   : BT*D bf16         =   8 MB
  //   hb   : BT*H bf16         =  32 MB   (reused per expert)
  char* ws = (char*)d_ws;
  float*          comb = (float*)ws;
  unsigned short* xb   = (unsigned short*)(ws + 131072);
  unsigned short* hb   = (unsigned short*)(ws + 131072 + (size_t)8 * 1024 * 1024);

  gate_kernel<<<BT / 256, 256, 0, stream>>>(x, gw, gb, comb);
  cvt_bf16_kernel<<<1024, 256, 0, stream>>>(x, xb, BT * D);

  for (int e = 0; e < E; ++e) {
    moe_gemm<D, H, true><<<dim3(H / 128, BT / 128), 256, 0, stream>>>(
        xb, w1 + (size_t)e * D * H, b1 + (size_t)e * H, hb, nullptr, nullptr, e);
    moe_gemm<H, D, false><<<dim3(D / 128, BT / 128), 256, 0, stream>>>(
        hb, w2 + (size_t)e * H * D, b2 + (size_t)e * D, nullptr, comb, out, e);
  }
}
